// VectorQuantizer_15968688407085
// MI455X (gfx1250) — matrix operand order repaired and checked by reading, not compiled, not run on hardware
//
#include <hip/hip_runtime.h>

typedef __attribute__((ext_vector_type(16))) _Float16 v16h;
typedef __attribute__((ext_vector_type(4)))  _Float16 v4h;
typedef __attribute__((ext_vector_type(8)))  float    v8f;
typedef __attribute__((ext_vector_type(4)))  float    v4f;

#define K_CB            1024
#define D_DIM           64
#define N_TOT           32768
#define WAVES_PER_BLOCK 8
#define ROWS_PER_WAVE   32                                   // 2 x 16-row A tiles
#define ROWS_PER_BLOCK  (WAVES_PER_BLOCK * ROWS_PER_WAVE)    // 256
#define NBLOCKS         (N_TOT / ROWS_PER_BLOCK)             // 128

__global__ void vq_init_loss(float* loss_ptr) {
    if (threadIdx.x == 0 && blockIdx.x == 0) *loss_ptr = 0.0f;
}

__global__ __launch_bounds__(256)
void vq_main_kernel(const float* __restrict__ x,
                    const float* __restrict__ cb,
                    float* __restrict__ out_q,
                    float* __restrict__ loss_ptr,
                    float* __restrict__ out_idx) {
    // 128 KB f16 codebook + 4 KB |e|^2 + small staging (<< 320 KB WGP LDS)
    __shared__ _Float16 lds_cb[K_CB * D_DIM];
    __shared__ float    lds_esq[K_CB];
    __shared__ int      lds_idx[ROWS_PER_BLOCK];
    __shared__ float    lds_loss[WAVES_PER_BLOCK];

    const int tid = threadIdx.x;

    // ---- Stage codebook -> LDS as f16 (f32x4 read -> packed f16x4 store) ----
    for (int i = tid; i < (K_CB * D_DIM) / 4; i += 256) {
        v4f v = ((const v4f*)cb)[i];
        v4h h;
        h.x = (_Float16)v.x; h.y = (_Float16)v.y;
        h.z = (_Float16)v.z; h.w = (_Float16)v.w;
        *(v4h*)&lds_cb[i * 4] = h;
    }
    // ---- |e_k|^2 in full f32 (vectorized) ----
    for (int k = tid; k < K_CB; k += 256) {
        const v4f* er = (const v4f*)(cb + (size_t)k * D_DIM);
        float s = 0.0f;
        #pragma unroll
        for (int c = 0; c < D_DIM / 4; ++c) {
            v4f e = er[c];
            s += e.x * e.x + e.y * e.y + e.z * e.z + e.w * e.w;
        }
        lds_esq[k] = s;
    }
    __syncthreads();

    const int wave = tid >> 5;
    const int lane = tid & 31;
    const int lo   = lane >> 4;      // half-wave select
    const int ln   = lane & 15;
    const int m0   = (blockIdx.x * WAVES_PER_BLOCK + wave) * ROWS_PER_WAVE;

    // ---- Load two A tiles (rows m0..m0+15 and m0+16..m0+31), K=64 each ----
    // f16 16x32 A layout: lane half lo holds K = lo*16..lo*16+15 (contiguous, matches B).
    v16h a0, a1, a2, a3;
    {
        const float* xr0 = x + (size_t)(m0 + ln) * D_DIM;
        const float* xr1 = x + (size_t)(m0 + 16 + ln) * D_DIM;
        #pragma unroll
        for (int j = 0; j < 8; ++j) {
            a0[j]     = (_Float16)xr0[      lo * 16 + j];
            a0[8 + j] = (_Float16)xr0[8   + lo * 16 + j];
            a1[j]     = (_Float16)xr0[32  + lo * 16 + j];
            a1[8 + j] = (_Float16)xr0[40  + lo * 16 + j];
            a2[j]     = (_Float16)xr1[      lo * 16 + j];
            a2[8 + j] = (_Float16)xr1[8   + lo * 16 + j];
            a3[j]     = (_Float16)xr1[32  + lo * 16 + j];
            a3[8 + j] = (_Float16)xr1[40  + lo * 16 + j];
        }
    }

    // ---- Running argmin of (|e|^2 - 2 x.e), two row-sets ----
    float bestA[8], bestB[8];
    int   bkA[8],   bkB[8];
    #pragma unroll
    for (int i = 0; i < 8; ++i) {
        bestA[i] = 3.4e38f; bkA[i] = 0;
        bestB[i] = 3.4e38f; bkB[i] = 0;
    }

    #pragma unroll 2
    for (int ct = 0; ct < K_CB / 16; ++ct) {
        const int n = ct * 16 + ln;  // codebook column this lane holds (B/C layout)
        // f16 32x16 B layout: lane half lo holds K = lo*16..lo*16+15 (contiguous).
        v16h b0 = *(const v16h*)&lds_cb[(size_t)n * D_DIM +  0 + lo * 16];
        v16h b1 = *(const v16h*)&lds_cb[(size_t)n * D_DIM + 32 + lo * 16];
        v8f accA = {}, accB = {};
        accA = __builtin_amdgcn_wmma_f32_16x16x32_f16(false, a0, false, b0,
                                                      (short)0, accA, false, false);
        accA = __builtin_amdgcn_wmma_f32_16x16x32_f16(false, a1, false, b1,
                                                      (short)0, accA, false, false);
        accB = __builtin_amdgcn_wmma_f32_16x16x32_f16(false, a2, false, b0,
                                                      (short)0, accB, false, false);
        accB = __builtin_amdgcn_wmma_f32_16x16x32_f16(false, a3, false, b1,
                                                      (short)0, accB, false, false);
        const float esq = lds_esq[n];
        #pragma unroll
        for (int i = 0; i < 8; ++i) {
            float dA = __builtin_fmaf(-2.0f, accA[i], esq);
            float dB = __builtin_fmaf(-2.0f, accB[i], esq);
            if (dA < bestA[i]) { bestA[i] = dA; bkA[i] = n; }
            if (dB < bestB[i]) { bestB[i] = dB; bkB[i] = n; }
        }
    }

    // ---- Argmin across the 16 lanes of each half (first-index tiebreak) ----
    #pragma unroll
    for (int m = 1; m <= 8; m <<= 1) {
        #pragma unroll
        for (int i = 0; i < 8; ++i) {
            float ov; int ok;
            ov = __shfl_xor(bestA[i], m, 32);
            ok = __shfl_xor(bkA[i],   m, 32);
            if (ov < bestA[i] || (ov == bestA[i] && ok < bkA[i])) {
                bestA[i] = ov; bkA[i] = ok;
            }
            ov = __shfl_xor(bestB[i], m, 32);
            ok = __shfl_xor(bkB[i],   m, 32);
            if (ov < bestB[i] || (ov == bestB[i] && ok < bkB[i])) {
                bestB[i] = ov; bkB[i] = ok;
            }
        }
    }

    // Element (vgpr v, half lo) is row M = v + 8*lo (+16 for set B); lane ln==v publishes.
    #pragma unroll
    for (int v = 0; v < 8; ++v) {
        if (ln == v) {
            int M0 = v + 8 * lo;
            int M1 = 16 + v + 8 * lo;
            lds_idx[wave * ROWS_PER_WAVE + M0] = bkA[v];
            lds_idx[wave * ROWS_PER_WAVE + M1] = bkB[v];
            out_idx[m0 + M0] = (float)bkA[v];
            out_idx[m0 + M1] = (float)bkB[v];
        }
    }
    __syncthreads();

    // ---- Quantize (exact f32 codebook gather, float4) + loss partials ----
    float part = 0.0f;
    const size_t blk_base = (size_t)blockIdx.x * ROWS_PER_BLOCK * D_DIM;
    for (int e = tid; e < (ROWS_PER_BLOCK * D_DIM) / 4; e += 256) {
        int r  = e >> 4;              // row within block (16 float4 per row)
        int c4 = e & 15;
        int gk = lds_idx[r];
        v4f q  = ((const v4f*)(cb + (size_t)gk * D_DIM))[c4];
        size_t g4 = (blk_base >> 2) + (size_t)e;   // float4 index into x / out_q
        v4f xi = ((const v4f*)x)[g4];
        ((v4f*)out_q)[g4] = q;        // STE forward value == quantized
        v4f d;
        d.x = q.x - xi.x; d.y = q.y - xi.y; d.z = q.z - xi.z; d.w = q.w - xi.w;
        part += d.x * d.x + d.y * d.y + d.z * d.z + d.w * d.w;
    }
    #pragma unroll
    for (int m = 16; m >= 1; m >>= 1) part += __shfl_xor(part, m, 32);
    if (lane == 0) lds_loss[wave] = part;
    __syncthreads();
    if (tid == 0) {
        float s = 0.0f;
        #pragma unroll
        for (int w = 0; w < WAVES_PER_BLOCK; ++w) s += lds_loss[w];
        // loss = q_latent + 0.25*e_latent, both numerically mean((q-x)^2)
        atomicAdd(loss_ptr, s * (1.25f / (float)(N_TOT * D_DIM)));
    }
}

extern "C" void kernel_launch(void* const* d_in, const int* in_sizes, int n_in,
                              void* d_out, int out_size, void* d_ws, size_t ws_size,
                              hipStream_t stream) {
    const float* x  = (const float*)d_in[0];   // inputs  [32,32,32,64] f32
    const float* cb = (const float*)d_in[1];   // codebook [1024,64] f32
    float* out      = (float*)d_out;

    float* out_q    = out;                                // 2,097,152 f32
    float* loss_ptr = out + (size_t)N_TOT * D_DIM;        // 1 f32
    float* out_idx  = out + (size_t)N_TOT * D_DIM + 1;    // 32768 (idx as f32)

    vq_init_loss<<<1, 32, 0, stream>>>(loss_ptr);
    vq_main_kernel<<<NBLOCKS, 256, 0, stream>>>(x, cb, out_q, loss_ptr, out_idx);
}
